// ReMoEMoE_72438918414738
// MI455X (gfx1250) — compile-verified, matrix operand order checked by reading, not run on hardware
//
#include <hip/hip_runtime.h>
#include <hip/hip_bf16.h>

// ---------------------------------------------------------------------------
// ReMoE fused MoE for MI455X (gfx1250, wave32, WMMA).
// Sizes fixed by the reference: B=4 T=2048 H=2048 E=8 F=1024, N = B*T = 8192.
// HBM-bound (~335MB/call -> ~14us floor @ 23.3 TB/s). Math in bf16 WMMA with
// fp32 accumulation; operands converted fp32->bf16 on the fly via HW cvt.
// hid intermediate kept bf16 in workspace (128MB) and streamed back to LDS
// with async global->LDS loads (ASYNCcnt path) in the down kernel.
// ---------------------------------------------------------------------------

#define N_TOK 8192
#define HDIM  2048
#define FDIM  1024
#define NEXP  8

typedef __attribute__((ext_vector_type(16))) __bf16 v16bf;
typedef __attribute__((ext_vector_type(8)))  __bf16 v8bf;
typedef __attribute__((ext_vector_type(8)))  float  v8f;

// 8x fp32 -> 8x bf16 (16B) via hardware conversion (v_cvt_pk_bf16_f32 class)
__device__ __forceinline__ uint4 pack8bf(const float* f) {
    union { v8bf v; uint4 q; } u;
#pragma unroll
    for (int i = 0; i < 8; ++i) u.v[i] = (__bf16)f[i];
    return u.q;
}

__device__ __forceinline__ unsigned short bfbits(float f) {
    union { __bf16 b; unsigned short s; } u;
    u.b = (__bf16)f;
    return u.s;
}

// Build a 16x32-bf16 WMMA fragment (8 VGPRs) from two contiguous 16B chunks.
__device__ __forceinline__ v16bf frag16(uint4 lo, uint4 hi) {
    union { uint4 q[2]; v16bf v; } u;
    u.q[0] = lo; u.q[1] = hi;
    return u.v;
}

// ---------------------------------------------------------------------------
// Kernel 1: router weights = relu(x @ W_router), one wave per token.
// ---------------------------------------------------------------------------
__global__ __launch_bounds__(256) void remoe_router(
    const float* __restrict__ x, const float* __restrict__ Wr,
    float* __restrict__ rw)
{
    const int wv   = threadIdx.x >> 5;
    const int lane = threadIdx.x & 31;
    const int token = blockIdx.x * 8 + wv;

    float acc[NEXP] = {0.f, 0.f, 0.f, 0.f, 0.f, 0.f, 0.f, 0.f};
    const float* xr = x + (size_t)token * HDIM;
    for (int h = lane; h < HDIM; h += 32) {
        float xv = xr[h];
        const float4* w4 = (const float4*)(Wr + (size_t)h * NEXP);
        float4 wa = w4[0], wb = w4[1];
        acc[0] += xv * wa.x; acc[1] += xv * wa.y;
        acc[2] += xv * wa.z; acc[3] += xv * wa.w;
        acc[4] += xv * wb.x; acc[5] += xv * wb.y;
        acc[6] += xv * wb.z; acc[7] += xv * wb.w;
    }
#pragma unroll
    for (int off = 16; off >= 1; off >>= 1)
#pragma unroll
        for (int e = 0; e < NEXP; ++e)
            acc[e] += __shfl_xor(acc[e], off, 32);

    if (lane < NEXP)
        rw[(size_t)token * NEXP + lane] = fmaxf(acc[lane], 0.0f);
}

// ---------------------------------------------------------------------------
// Kernel 2: hid[e] = silu(x @ Wg[e]) * (x @ Wu[e]), stored bf16 in workspace.
// Block tile: 64 tokens x 64 F-cols; 8 waves, each a 16x32 output tile
// (2 gate + 2 up f32 accumulators). K-loop over H in steps of 32.
// ---------------------------------------------------------------------------
__global__ __launch_bounds__(256) void remoe_gateup(
    const float* __restrict__ x, const float* __restrict__ Wg,
    const float* __restrict__ Wu, unsigned short* __restrict__ hid)
{
    __shared__ unsigned short Xs[64][40];   // [token][k] bf16 (80B rows, 16B aligned)
    __shared__ unsigned short Gt[64][40];   // W_gate tile transposed: [f][k]
    __shared__ unsigned short Ut[64][40];   // W_up   tile transposed: [f][k]

    const int t0 = blockIdx.x * 64;
    const int f0 = blockIdx.y * 64;
    const int e  = blockIdx.z;
    const float* Wge = Wg + (size_t)e * HDIM * FDIM;
    const float* Wue = Wu + (size_t)e * HDIM * FDIM;

    const int tid  = threadIdx.x;
    const int lane = tid & 31;
    const int wv   = tid >> 5;
    const int half = lane >> 4;       // lanes 16-31 hold k 8-15 and 24-31
    const int lcol = lane & 15;
    const int mB   = 16 * (wv & 3);   // wave's token-row base in the tile
    const int nB   = 32 * (wv >> 2);  // wave's F-col base in the tile

    v8f g0 = {}, g1 = {}, u0 = {}, u1 = {};

    const int xr = tid >> 2, xc = (tid & 3) * 8;  // X staging: row-major b128
    const int wCol = tid & 63;                    // W staging: one column / thread
    const int wK   = (tid >> 6) * 8;              // 8-deep k-chunk (coalesced per j)

    for (int k0 = 0; k0 < HDIM; k0 += 32) {
        __syncthreads();
        // ---- stage X tile [64 tok][32 k] fp32 -> bf16, row-major ----
        {
            const float* p = x + (size_t)(t0 + xr) * HDIM + k0 + xc;
            float4 a = ((const float4*)p)[0];
            float4 b = ((const float4*)p)[1];
            float v[8] = {a.x, a.y, a.z, a.w, b.x, b.y, b.z, b.w};
            *(uint4*)&Xs[xr][xc] = pack8bf(v);
        }
        // ---- stage Wg/Wu tiles [32 k][64 f] directly transposed: each thread
        //      owns column wCol, k-chunk wK..wK+7; per-j loads are 32
        //      lane-consecutive floats (fully coalesced). One b128 LDS store.
        {
            const float* pg = Wge + (size_t)(k0 + wK) * FDIM + f0 + wCol;
            const float* pu = Wue + (size_t)(k0 + wK) * FDIM + f0 + wCol;
            float gv[8], uv[8];
#pragma unroll
            for (int j = 0; j < 8; ++j) {
                gv[j] = pg[(size_t)j * FDIM];
                uv[j] = pu[(size_t)j * FDIM];
            }
            *(uint4*)&Gt[wCol][wK] = pack8bf(gv);
            *(uint4*)&Ut[wCol][wK] = pack8bf(uv);
        }
        if (k0 + 32 < HDIM) {
            __builtin_prefetch(x   + (size_t)(t0 + xr) * HDIM + k0 + 32 + xc, 0, 1);
            __builtin_prefetch(Wge + (size_t)(k0 + 32 + wK) * FDIM + f0 + wCol, 0, 1);
            __builtin_prefetch(Wue + (size_t)(k0 + 32 + wK) * FDIM + f0 + wCol, 0, 1);
        }
        __syncthreads();

        // ---- fragments (ISA 16-bit A layout: two contiguous 8-elem chunks) ----
        const int am = mB + lcol;
        v16bf A = frag16(*(const uint4*)&Xs[am][8 * half],
                         *(const uint4*)&Xs[am][16 + 8 * half]);
        const int bn0 = nB + lcol, bn1 = bn0 + 16;
        v16bf Bg0 = frag16(*(const uint4*)&Gt[bn0][8 * half],
                           *(const uint4*)&Gt[bn0][16 + 8 * half]);
        v16bf Bg1 = frag16(*(const uint4*)&Gt[bn1][8 * half],
                           *(const uint4*)&Gt[bn1][16 + 8 * half]);
        v16bf Bu0 = frag16(*(const uint4*)&Ut[bn0][8 * half],
                           *(const uint4*)&Ut[bn0][16 + 8 * half]);
        v16bf Bu1 = frag16(*(const uint4*)&Ut[bn1][8 * half],
                           *(const uint4*)&Ut[bn1][16 + 8 * half]);

        g0 = __builtin_amdgcn_wmma_f32_16x16x32_bf16(false, A, false, Bg0, (short)0, g0, false, false);
        g1 = __builtin_amdgcn_wmma_f32_16x16x32_bf16(false, A, false, Bg1, (short)0, g1, false, false);
        u0 = __builtin_amdgcn_wmma_f32_16x16x32_bf16(false, A, false, Bu0, (short)0, u0, false, false);
        u1 = __builtin_amdgcn_wmma_f32_16x16x32_bf16(false, A, false, Bu1, (short)0, u1, false, false);
    }

    // ---- fused SiLU(gate)*up, store bf16 ----
    unsigned short* hb = hid + (size_t)e * N_TOK * FDIM;
#pragma unroll
    for (int r = 0; r < 8; ++r) {
        const int row = t0 + mB + r + 8 * half;    // C/D layout: M = r + 8*(lane/16)
        float sg0 = g0[r] / (1.0f + __expf(-g0[r]));
        float sg1 = g1[r] / (1.0f + __expf(-g1[r]));
        hb[(size_t)row * FDIM + f0 + nB + lcol]      = bfbits(sg0 * u0[r]);
        hb[(size_t)row * FDIM + f0 + nB + 16 + lcol] = bfbits(sg1 * u1[r]);
    }
}

// ---------------------------------------------------------------------------
// Kernel 3: out[n,h] = sum_e rw[n,e] * (hid[e] @ Wd[e])[n,h]
// Block tile: 64 tokens x 64 H-cols; 8 waves, each 16x32.
// hid tile (already bf16 in global) is brought into LDS with
// GLOBAL_LOAD_ASYNC_TO_LDS_B128 (ASYNCcnt), overlapped with W_down staging.
// ---------------------------------------------------------------------------
__global__ __launch_bounds__(256) void remoe_down(
    const unsigned short* __restrict__ hid, const float* __restrict__ Wd,
    const float* __restrict__ rw, float* __restrict__ out)
{
    __shared__ unsigned short Hs[64][40];   // hid tile [token][f] bf16
    __shared__ unsigned short Dt[64][40];   // W_down tile transposed [h][f]
    __shared__ float          Rs[64][NEXP]; // router weights for this token tile

    const int t0 = blockIdx.x * 64;
    const int h0 = blockIdx.y * 64;

    const int tid  = threadIdx.x;
    const int lane = tid & 31;
    const int wv   = tid >> 5;
    const int half = lane >> 4;
    const int lcol = lane & 15;
    const int mB   = 16 * (wv & 3);
    const int nB   = 32 * (wv >> 2);

    for (int i = tid; i < 64 * NEXP; i += 256)
        Rs[i >> 3][i & 7] = rw[(size_t)(t0 + (i >> 3)) * NEXP + (i & 7)];

    const int xr = tid >> 2, xc = (tid & 3) * 8;  // hid staging (b128 per thread)
    const int wCol = tid & 63;                    // W_down: one column / thread
    const int wK   = (tid >> 6) * 8;

    const unsigned hsDst = (unsigned)(uintptr_t)&Hs[xr][xc]; // LDS byte offset

    v8f o0 = {}, o1 = {};

    for (int e = 0; e < NEXP; ++e) {
        const unsigned short* he = hid + (size_t)e * N_TOK * FDIM;
        const float*          We = Wd  + (size_t)e * FDIM * HDIM;
        v8f a0 = {}, a1 = {};

        for (int k0 = 0; k0 < FDIM; k0 += 32) {
            __syncthreads();
            // async global->LDS fill of the bf16 hid tile (16B per lane)
            {
                unsigned long long ga =
                    (unsigned long long)(he + (size_t)(t0 + xr) * FDIM + k0 + xc);
                asm volatile("global_load_async_to_lds_b128 %0, %1, off"
                             :: "v"(hsDst), "v"(ga) : "memory");
            }
            // overlap: stage W_down tile [32 f][64 h] fp32 -> bf16, transposed
            {
                const float* p = We + (size_t)(k0 + wK) * HDIM + h0 + wCol;
                float v[8];
#pragma unroll
                for (int j = 0; j < 8; ++j) v[j] = p[(size_t)j * HDIM];
                *(uint4*)&Dt[wCol][wK] = pack8bf(v);
            }
            if (k0 + 32 < FDIM)
                __builtin_prefetch(We + (size_t)(k0 + 32 + wK) * HDIM + h0 + wCol, 0, 1);
            asm volatile("s_wait_asynccnt 0" ::: "memory");
            __syncthreads();

            const int am = mB + lcol;
            v16bf A = frag16(*(const uint4*)&Hs[am][8 * half],
                             *(const uint4*)&Hs[am][16 + 8 * half]);
            const int bn0 = nB + lcol, bn1 = bn0 + 16;
            v16bf B0 = frag16(*(const uint4*)&Dt[bn0][8 * half],
                              *(const uint4*)&Dt[bn0][16 + 8 * half]);
            v16bf B1 = frag16(*(const uint4*)&Dt[bn1][8 * half],
                              *(const uint4*)&Dt[bn1][16 + 8 * half]);

            a0 = __builtin_amdgcn_wmma_f32_16x16x32_bf16(false, A, false, B0, (short)0, a0, false, false);
            a1 = __builtin_amdgcn_wmma_f32_16x16x32_bf16(false, A, false, B1, (short)0, a1, false, false);
        }

        // weighted accumulation (per-row router weight)
#pragma unroll
        for (int r = 0; r < 8; ++r) {
            float w = Rs[mB + r + 8 * half][e];
            o0[r] += w * a0[r];
            o1[r] += w * a1[r];
        }
    }

#pragma unroll
    for (int r = 0; r < 8; ++r) {
        const int row = t0 + mB + r + 8 * half;
        out[(size_t)row * HDIM + h0 + nB + lcol]      = o0[r];
        out[(size_t)row * HDIM + h0 + nB + 16 + lcol] = o1[r];
    }
}

// ---------------------------------------------------------------------------
extern "C" void kernel_launch(void* const* d_in, const int* in_sizes, int n_in,
                              void* d_out, int out_size, void* d_ws, size_t ws_size,
                              hipStream_t stream) {
    const float* x  = (const float*)d_in[0];   // [N, H]
    const float* Wr = (const float*)d_in[1];   // [H, E]
    const float* Wg = (const float*)d_in[2];   // [E, H, F]
    const float* Wu = (const float*)d_in[3];   // [E, H, F]
    const float* Wd = (const float*)d_in[4];   // [E, F, H]

    float* out = (float*)d_out;                        // [N, H]
    float* rw  = out + (size_t)N_TOK * HDIM;           // [N, E] (2nd tuple output)
    unsigned short* hid = (unsigned short*)d_ws;       // [E, N, F] bf16 (128 MB)

    remoe_router<<<N_TOK / 8, 256, 0, stream>>>(x, Wr, rw);
    remoe_gateup<<<dim3(N_TOK / 64, FDIM / 64, NEXP), 256, 0, stream>>>(x, Wg, Wu, hid);
    remoe_down<<<dim3(N_TOK / 64, HDIM / 64), 256, 0, stream>>>(hid, Wd, rw, out);
}
